// sLSTMblock_77919296684765
// MI455X (gfx1250) — matrix operand order repaired and checked by reading, not compiled, not run on hardware
//
#include <hip/hip_runtime.h>
#include <math.h>

// ---------------------------------------------------------------------------
// sLSTM block, MI455X (gfx1250, wave32, WMMA).
// Heavy op: block-diagonal 4-gate projection -> v_wmma_f32_16x16x32_f16
// (f16 A/B, f32 accumulate; K=1024 per output). Everything else is
// bandwidth-bound LN/pointwise streaming with deterministic reductions.
// Recurrent Wr_gates term is exactly zero (ht_1 == 0) and is skipped.
// ---------------------------------------------------------------------------

typedef __attribute__((ext_vector_type(16))) _Float16 v16h;
typedef __attribute__((ext_vector_type(8)))  _Float16 v8h;
typedef __attribute__((ext_vector_type(8)))  float    v8f;

#define DM     4096      // model dim
#define TOKS   8192      // 4 * 2048 rows
#define IB     1024      // block size (DM / 4)
#define NGATE  4
#define NBLK   4
#define DFFN   5461
#define LNEPS  1e-5f
#define WGS    256       // row-stage workgroups
#define TPB    (TOKS / WGS)   // 32 tokens per workgroup

// ---- workspace layout (bytes); total ~785 MB -------------------------------
static constexpr size_t OFF_XB  = 0;                                   // f16 silu(x)   [TOKS][DM]
static constexpr size_t OFF_WB  = OFF_XB  + (size_t)TOKS * DM * 2;     // f16 W_gates   [4][4][IB][IB]
static constexpr size_t OFF_PRE = OFF_WB  + (size_t)NGATE*NBLK*IB*IB*2;// f32 pre       [4][TOKS][DM]
static constexpr size_t OFF_O   = OFF_PRE + (size_t)NGATE*TOKS*DM*4;   // f32 o-gate    [TOKS][DM]
static constexpr size_t OFF_CTP = OFF_O   + (size_t)TOKS*DM*4;         // f32 ct partials [WGS][DM]
static constexpr size_t OFF_NTP = OFF_CTP + (size_t)WGS*DM*4;          // f32 nt partials [WGS][DM]
static constexpr size_t OFF_HTP = OFF_NTP + (size_t)WGS*DM*4;          // f32 ht partials [WGS][DM]
static constexpr size_t OFF_CTS = OFF_HTP + (size_t)WGS*DM*4;          // f32 ct_sum [DM]
static constexpr size_t OFF_NTS = OFF_CTS + (size_t)DM*4;              // f32 nt_sum [DM]
static constexpr size_t OFF_HTS = OFF_NTS + (size_t)DM*4;              // f32 ht_sum [DM]
static constexpr size_t OFF_SL  = OFF_HTS + (size_t)DM*4;              // f32 slstm_out [DM]
static constexpr size_t OFF_LR  = OFF_SL  + (size_t)DM*4;              // f32 left*right [DFFN]
static constexpr size_t OFF_ST  = OFF_LR  + (((size_t)DFFN*4 + 15) & ~(size_t)15); // f32 stats[2]

// ---------------------------------------------------------------------------
__device__ __forceinline__ float block_sum_256(float v, float* red) {
    const int t = threadIdx.x;
    red[t] = v; __syncthreads();
    #pragma unroll
    for (int s = 128; s > 0; s >>= 1) {
        if (t < s) red[t] += red[t + s];
        __syncthreads();
    }
    float r = red[0]; __syncthreads();
    return r;
}

// ---- f16 staging -----------------------------------------------------------
__global__ void silu_to_f16_kernel(const float* __restrict__ x,
                                   _Float16* __restrict__ xb, int n) {
    int i = blockIdx.x * blockDim.x + threadIdx.x;
    if (i < n) {
        float v = x[i];
        xb[i] = (_Float16)(v / (1.f + __expf(-v)));   // silu
    }
}

__global__ void cvt_f16_kernel(const float* __restrict__ w,
                               _Float16* __restrict__ wb, int n) {
    int i = blockIdx.x * blockDim.x + threadIdx.x;
    if (i < n) wb[i] = (_Float16)w[i];
}

// ---- stage 1: block-diagonal gate GEMM via WMMA ----------------------------
// grid: (TOKS/32, IB/64, NGATE*NBLK), block: 256 (8 waves, 2x4 tile grid)
// pre[g][t][kb*IB + c] = sum_i silu(x)[t][kb*IB+i] * W_gates[g][kb][c][i] + b_gates[g][...]
__global__ void gate_gemm_wmma(const _Float16* __restrict__ xb,
                               const _Float16* __restrict__ wb,
                               const float* __restrict__ bg,
                               float* __restrict__ pre) {
    const int lane   = threadIdx.x & 31;
    const int wave   = threadIdx.x >> 5;   // 0..7
    const int mT     = wave >> 2;          // 0..1 (token tile)
    const int nT     = wave & 3;           // 0..3 (col tile)
    const int half16 = lane >> 4;          // 0/1 lane group
    const int r16    = lane & 15;

    const int tokBase = blockIdx.x * 32 + mT * 16;
    const int colBase = blockIdx.y * 64 + nT * 16;      // within the 1024 block
    const int kb      = blockIdx.z & 3;
    const int g       = blockIdx.z >> 2;

    // A: row (tokBase+r16) of silu(x), K slice = columns of block kb.
    // ISA A layout: lanes 0-15 hold K {0..7,16..23}, lanes 16-31 K {8..15,24..31}.
    const _Float16* Arow = xb + (size_t)(tokBase + r16) * DM + (size_t)kb * IB;
    const int kaoff = half16 * 8;

    // B[k][n] = W[g][kb][colBase+n][k]; lane n holds K {0..7,16..23}, lane n+16 K {8..15,24..31}.
    const _Float16* Brow = wb + ((((size_t)g * NBLK + kb) * IB) + (colBase + r16)) * IB;
    const int kboff = half16 * 8;

    v8f acc = {};
    #pragma unroll 4
    for (int k = 0; k < IB; k += 32) {
        v8h a0 = *(const v8h*)(Arow + k + kaoff);
        v8h a1 = *(const v8h*)(Arow + k + kaoff + 16);
        v8h b0 = *(const v8h*)(Brow + k + kboff);
        v8h b1 = *(const v8h*)(Brow + k + kboff + 16);
        v16h a, b;
        #pragma unroll
        for (int e = 0; e < 8; ++e) {
            a[e] = a0[e]; a[e + 8] = a1[e];
            b[e] = b0[e]; b[e + 8] = b1[e];
        }
        acc = __builtin_amdgcn_wmma_f32_16x16x32_f16(
                  false, a, false, b, (short)0, acc, false, false);
    }

    // C/D layout: lanes 0-15: N=lane, M=vgpr r; lanes 16-31: N=lane-16, M=8+r.
    const int dcol = kb * IB + colBase + r16;
    const float bias = bg[g * DM + dcol];
    float* outp = pre + (((size_t)g * TOKS + tokBase + half16 * 8) * DM) + dcol;
    #pragma unroll
    for (int rr = 0; rr < 8; ++rr)
        outp[(size_t)rr * DM] = acc[rr] + bias;
}

// ---- stage 2: per-row LN + activations; per-WG partial column sums ---------
// grid WGS, block 256; each thread owns 16 columns (c = tid + 256*j).
__global__ void gate_rows_kernel(const float* __restrict__ pre,
                                 const float* __restrict__ lg,
                                 const float* __restrict__ lb,
                                 float* __restrict__ o_ws,
                                 float* __restrict__ ct_part,
                                 float* __restrict__ nt_part) {
    __shared__ float red[256];
    const int tid  = threadIdx.x;
    const int tok0 = blockIdx.x * TPB;
    const float inv = 1.f / (float)DM;

    float acc_ct[16], acc_nt[16];
    #pragma unroll
    for (int j = 0; j < 16; ++j) { acc_ct[j] = 0.f; acc_nt[j] = 0.f; }

    for (int t = tok0; t < tok0 + TPB; ++t) {
        const float* p0 = pre + (size_t)t * DM;
        const float* p1 = p0 + (size_t)1 * TOKS * DM;
        const float* p2 = p0 + (size_t)2 * TOKS * DM;
        const float* p3 = p0 + (size_t)3 * TOKS * DM;

        float a0[16], a1[16], iv[16], cv[16];
        float s, q;

        // gate 0 (log i) and gate 1 (log f): row LNs
        s = 0.f; q = 0.f;
        #pragma unroll
        for (int j = 0; j < 16; ++j) { float v = p0[tid + j*256]; a0[j] = v; s += v; q += v*v; }
        float mu0 = block_sum_256(s, red) * inv;
        float rs0 = rsqrtf(block_sum_256(q, red) * inv - mu0*mu0 + LNEPS);

        s = 0.f; q = 0.f;
        #pragma unroll
        for (int j = 0; j < 16; ++j) { float v = p1[tid + j*256]; a1[j] = v; s += v; q += v*v; }
        float mu1 = block_sum_256(s, red) * inv;
        float rs1 = rsqrtf(block_sum_256(q, red) * inv - mu1*mu1 + LNEPS);

        #pragma unroll
        for (int j = 0; j < 16; ++j) {
            int c = tid + j*256;
            float li = (a0[j] - mu0) * rs0 * lg[0*DM + c] + lb[0*DM + c];
            float lf = (a1[j] - mu1) * rs1 * lg[1*DM + c] + lb[1*DM + c];
            float m  = fmaxf(li, lf);            // stabilizer (mt_1 = 0)
            iv[j] = __expf(li - m);              // i gate (f is dead code in ref)
        }

        // gate 2 -> o = sigmoid(LN(pre2)), spill to ws for stage 3
        s = 0.f; q = 0.f;
        #pragma unroll
        for (int j = 0; j < 16; ++j) { float v = p2[tid + j*256]; a0[j] = v; s += v; q += v*v; }
        float mu2 = block_sum_256(s, red) * inv;
        float rs2 = rsqrtf(block_sum_256(q, red) * inv - mu2*mu2 + LNEPS);
        #pragma unroll
        for (int j = 0; j < 16; ++j) {
            int c = tid + j*256;
            float lo = (a0[j] - mu2) * rs2 * lg[2*DM + c] + lb[2*DM + c];
            o_ws[(size_t)t * DM + c] = 1.f / (1.f + __expf(-lo));
        }

        // gate 3 -> z = tanh(LN(pre3)); ct = i*z
        s = 0.f; q = 0.f;
        #pragma unroll
        for (int j = 0; j < 16; ++j) { float v = p3[tid + j*256]; a1[j] = v; s += v; q += v*v; }
        float mu3 = block_sum_256(s, red) * inv;
        float rs3 = rsqrtf(block_sum_256(q, red) * inv - mu3*mu3 + LNEPS);
        #pragma unroll
        for (int j = 0; j < 16; ++j) {
            int c = tid + j*256;
            float z = tanhf((a1[j] - mu3) * rs3 * lg[3*DM + c] + lb[3*DM + c]);
            cv[j] = iv[j] * z;
        }

        // accumulate LN(ct) (gamma/beta #4) and LN(i) (#5) row-wise
        s = 0.f; q = 0.f;
        #pragma unroll
        for (int j = 0; j < 16; ++j) { s += cv[j]; q += cv[j]*cv[j]; }
        float mu4 = block_sum_256(s, red) * inv;
        float rs4 = rsqrtf(block_sum_256(q, red) * inv - mu4*mu4 + LNEPS);
        #pragma unroll
        for (int j = 0; j < 16; ++j) {
            int c = tid + j*256;
            acc_ct[j] += (cv[j] - mu4) * rs4 * lg[4*DM + c] + lb[4*DM + c];
        }

        s = 0.f; q = 0.f;
        #pragma unroll
        for (int j = 0; j < 16; ++j) { s += iv[j]; q += iv[j]*iv[j]; }
        float mu5 = block_sum_256(s, red) * inv;
        float rs5 = rsqrtf(block_sum_256(q, red) * inv - mu5*mu5 + LNEPS);
        #pragma unroll
        for (int j = 0; j < 16; ++j) {
            int c = tid + j*256;
            acc_nt[j] += (iv[j] - mu5) * rs5 * lg[5*DM + c] + lb[5*DM + c];
        }
    }

    #pragma unroll
    for (int j = 0; j < 16; ++j) {
        int c = tid + j*256;
        ct_part[(size_t)blockIdx.x * DM + c] = acc_ct[j];
        nt_part[(size_t)blockIdx.x * DM + c] = acc_nt[j];
    }
}

// ---- fixed-order column reduction: [WGS][DM] -> [DM] (deterministic) -------
__global__ void col_reduce_kernel(const float* __restrict__ part,
                                  float* __restrict__ out) {
    int d = blockIdx.x * blockDim.x + threadIdx.x;
    float s = 0.f;
    for (int w = 0; w < WGS; ++w) s += part[(size_t)w * DM + d];
    out[d] = s;
}

// ---- stage 3: ht rows; r = ct_sum/nt_sum (mean count cancels) --------------
__global__ void ht_rows_kernel(const float* __restrict__ o_ws,
                               const float* __restrict__ ct_sum,
                               const float* __restrict__ nt_sum,
                               const float* __restrict__ lg,
                               const float* __restrict__ lb,
                               float* __restrict__ ht_part) {
    __shared__ float red[256];
    const int tid  = threadIdx.x;
    const int tok0 = blockIdx.x * TPB;
    const float inv = 1.f / (float)DM;

    float rloc[16], acc[16];
    #pragma unroll
    for (int j = 0; j < 16; ++j) {
        int c = tid + j*256;
        rloc[j] = ct_sum[c] / nt_sum[c];
        acc[j] = 0.f;
    }

    for (int t = tok0; t < tok0 + TPB; ++t) {
        float h[16], s = 0.f, q = 0.f;
        #pragma unroll
        for (int j = 0; j < 16; ++j) {
            int c = tid + j*256;
            h[j] = o_ws[(size_t)t * DM + c] * rloc[j];
            s += h[j]; q += h[j]*h[j];
        }
        float mu = block_sum_256(s, red) * inv;
        float rs = rsqrtf(block_sum_256(q, red) * inv - mu*mu + LNEPS);
        #pragma unroll
        for (int j = 0; j < 16; ++j) {
            int c = tid + j*256;
            acc[j] += (h[j] - mu) * rs * lg[6*DM + c] + lb[6*DM + c];
        }
    }
    #pragma unroll
    for (int j = 0; j < 16; ++j)
        ht_part[(size_t)blockIdx.x * DM + tid + j*256] = acc[j];
}

// ---- stage 4: slstm_out = LN(ht_mean) (single block) -----------------------
__global__ void slstm_kernel(const float* __restrict__ ht_sum,
                             const float* __restrict__ lg,
                             const float* __restrict__ lb,
                             float* __restrict__ slstm) {
    __shared__ float red[256];
    const int tid = threadIdx.x;
    const float inv = 1.f / (float)DM;
    float h[16], s = 0.f, q = 0.f;
    #pragma unroll
    for (int j = 0; j < 16; ++j) {
        h[j] = ht_sum[tid + j*256] * (1.f / (float)TOKS);
        s += h[j]; q += h[j]*h[j];
    }
    float mu = block_sum_256(s, red) * inv;
    float rs = rsqrtf(block_sum_256(q, red) * inv - mu*mu + LNEPS);
    #pragma unroll
    for (int j = 0; j < 16; ++j) {
        int c = tid + j*256;
        slstm[c] = (h[j] - mu) * rs * lg[7*DM + c] + lb[7*DM + c];
    }
}

// ---- MLP tail: left/right matvecs + exact gelu -----------------------------
__global__ void leftright_kernel(const float* __restrict__ slstm,
                                 const float* __restrict__ Wl, const float* __restrict__ bl,
                                 const float* __restrict__ Wr, const float* __restrict__ br,
                                 float* __restrict__ lr) {
    __shared__ float red[256];
    const int j = blockIdx.x;
    float sl = 0.f, sr = 0.f;
    for (int i = threadIdx.x; i < DM; i += 256) {
        float sv = slstm[i];
        sl += sv * Wl[(size_t)j * DM + i];
        sr += sv * Wr[(size_t)j * DM + i];
    }
    float SL = block_sum_256(sl, red);
    float SR = block_sum_256(sr, red);
    if (threadIdx.x == 0) {
        float left = SL + bl[j];
        float xr   = SR + br[j];
        float right = 0.5f * xr * (1.f + erff(xr * 0.70710678118654752f)); // exact gelu
        lr[j] = left * right;
    }
}

__global__ void lr_stats_kernel(const float* __restrict__ lr,
                                float* __restrict__ stats) {
    __shared__ float red[256];
    float s = 0.f, q = 0.f;
    for (int j = threadIdx.x; j < DFFN; j += 256) {
        float v = lr[j]; s += v; q += v*v;
    }
    float S = block_sum_256(s, red);
    float Q = block_sum_256(q, red);
    if (threadIdx.x == 0) { stats[0] = S; stats[1] = Q; }
}

__global__ void proj_kernel(const float* __restrict__ lr,
                            const float* __restrict__ stats,
                            const float* __restrict__ g, const float* __restrict__ b,
                            const float* __restrict__ Wp, const float* __restrict__ bp,
                            float* __restrict__ out) {
    __shared__ float red[256];
    const float mean = stats[0] * (1.f / (float)DFFN);
    const float var  = stats[1] * (1.f / (float)DFFN) - mean * mean;
    const float rs   = rsqrtf(var + LNEPS);
    const int d = blockIdx.x;
    float s = 0.f;
    for (int j = threadIdx.x; j < DFFN; j += 256) {
        float v = (lr[j] - mean) * rs * g[j] + b[j];
        s += v * Wp[(size_t)d * DFFN + j];
    }
    float S = block_sum_256(s, red);
    if (threadIdx.x == 0) out[d] = S + bp[d];
}

// ---------------------------------------------------------------------------
extern "C" void kernel_launch(void* const* d_in, const int* in_sizes, int n_in,
                              void* d_out, int out_size, void* d_ws, size_t ws_size,
                              hipStream_t stream) {
    (void)in_sizes; (void)n_in; (void)out_size; (void)ws_size;
    const float* x        = (const float*)d_in[0];
    const float* W_gates  = (const float*)d_in[1];
    const float* b_gates  = (const float*)d_in[2];
    /* d_in[3] = Wr_gates: multiplied by zero initial state -> unused */
    const float* ln_gamma = (const float*)d_in[4];
    const float* ln_beta  = (const float*)d_in[5];
    const float* lnout_g  = (const float*)d_in[6];
    const float* lnout_b  = (const float*)d_in[7];
    const float* W_left   = (const float*)d_in[8];
    const float* b_left   = (const float*)d_in[9];
    const float* W_right  = (const float*)d_in[10];
    const float* b_right  = (const float*)d_in[11];
    const float* W_proj   = (const float*)d_in[12];
    const float* b_proj   = (const float*)d_in[13];
    float* out = (float*)d_out;

    char* ws = (char*)d_ws;
    _Float16* xb    = (_Float16*)(ws + OFF_XB);
    _Float16* wb    = (_Float16*)(ws + OFF_WB);
    float* pre      = (float*)(ws + OFF_PRE);
    float* o_ws     = (float*)(ws + OFF_O);
    float* ct_part  = (float*)(ws + OFF_CTP);
    float* nt_part  = (float*)(ws + OFF_NTP);
    float* ht_part  = (float*)(ws + OFF_HTP);
    float* ct_sum   = (float*)(ws + OFF_CTS);
    float* nt_sum   = (float*)(ws + OFF_NTS);
    float* ht_sum   = (float*)(ws + OFF_HTS);
    float* slstm    = (float*)(ws + OFF_SL);
    float* lr       = (float*)(ws + OFF_LR);
    float* stats    = (float*)(ws + OFF_ST);

    // 1) stage f16 operands (one-shot silu + convert)
    {
        int n = TOKS * DM;            // 33,554,432
        silu_to_f16_kernel<<<(n + 255) / 256, 256, 0, stream>>>(x, xb, n);
        int nw = NGATE * NBLK * IB * IB;  // 16,777,216
        cvt_f16_kernel<<<(nw + 255) / 256, 256, 0, stream>>>(W_gates, wb, nw);
    }

    // 2) WMMA gate GEMM: 275 GFLOP f16->f32
    {
        dim3 grid(TOKS / 32, IB / 64, NGATE * NBLK);
        gate_gemm_wmma<<<grid, 256, 0, stream>>>(xb, wb, b_gates, pre);
    }

    // 3) row LNs + activations -> o, per-WG partial sums for ct/nt
    gate_rows_kernel<<<WGS, 256, 0, stream>>>(pre, ln_gamma, ln_beta,
                                              o_ws, ct_part, nt_part);
    col_reduce_kernel<<<DM / 256, 256, 0, stream>>>(ct_part, ct_sum);
    col_reduce_kernel<<<DM / 256, 256, 0, stream>>>(nt_part, nt_sum);

    // 4) ht rows with r = ct_sum/nt_sum, then reduce
    ht_rows_kernel<<<WGS, 256, 0, stream>>>(o_ws, ct_sum, nt_sum,
                                            ln_gamma, ln_beta, ht_part);
    col_reduce_kernel<<<DM / 256, 256, 0, stream>>>(ht_part, ht_sum);

    // 5) slstm_out = LN(mean(ht rows))
    slstm_kernel<<<1, 256, 0, stream>>>(ht_sum, ln_gamma, ln_beta, slstm);

    // 6) MLP tail
    leftright_kernel<<<DFFN, 256, 0, stream>>>(slstm, W_left, b_left,
                                               W_right, b_right, lr);
    lr_stats_kernel<<<1, 256, 0, stream>>>(lr, stats);
    proj_kernel<<<DM, 256, 0, stream>>>(lr, stats, lnout_g, lnout_b,
                                        W_proj, b_proj, out);
}
